// FusedMoEBlockedF8_12214886989885
// MI455X (gfx1250) — matrix operand order repaired and checked by reading, not compiled, not run on hardware
//
#include <hip/hip_runtime.h>
#include <hip/hip_bf16.h>

// ---------------- problem constants (from reference) ----------------
constexpr int T    = 2048;   // tokens
constexpr int H    = 2048;   // hidden
constexpr int F    = 768;    // intermediate
constexpr int E    = 8;      // experts
constexpr int TOPK = 2;
constexpr int BLK  = 128;    // quant block
constexpr int CAP  = T * TOPK;          // worst-case pairs per expert
constexpr int MROWS = 32;               // tokens per workgroup (2 M-tiles)
constexpr int TILES_PER_E = CAP / MROWS;

// ---------------- WMMA vector types ----------------
typedef __attribute__((ext_vector_type(16))) __bf16 v16bf;
typedef __attribute__((ext_vector_type(8)))  float  v8f;
typedef __attribute__((ext_vector_type(2)))  __bf16 v2bf;

union Frag {
    v16bf v;
    unsigned u[8];
    uint4 q[2];
};

// two f32 -> packed bf16 dword (native v_cvt_pk_bf16_f32) -- memory-layout path
static __device__ __forceinline__ unsigned cvt2(float lo, float hi) {
#if __has_builtin(__builtin_amdgcn_cvt_pk_bf16_f32)
    union { v2bf v; unsigned u; } c;
    c.v = __builtin_amdgcn_cvt_pk_bf16_f32(lo, hi);
    return c.u;
#else
    union { __bf16 h[2]; unsigned u; } c;
    c.h[0] = (__bf16)lo; c.h[1] = (__bf16)hi;
    return c.u;
#endif
}
static __device__ __forceinline__ unsigned short f2bf(float x) {
    union { __bf16 h; unsigned short s; } c;
    c.h = (__bf16)x;
    return c.s;
}

// ---------------- kernel 0: zero output + counters ----------------
__global__ void k_zero(float* __restrict__ out, int nOut, int* __restrict__ cnt) {
    int i = blockIdx.x * blockDim.x + threadIdx.x;
    if (i < nOut) out[i] = 0.0f;
    if (i < E)    cnt[i] = 0;
}

// ---------------- kernel 1: routing ----------------
__global__ void k_route(const float* __restrict__ tw, const int* __restrict__ ids,
                        int* __restrict__ cnt, int* __restrict__ pairlist,
                        float* __restrict__ twn) {
    int t = blockIdx.x * blockDim.x + threadIdx.x;
    if (t >= T) return;
    float w0 = tw[t * TOPK + 0];
    float w1 = tw[t * TOPK + 1];
    float inv = 1.0f / (w0 + w1);
#pragma unroll
    for (int k = 0; k < TOPK; ++k) {
        int p = t * TOPK + k;
        twn[p] = (k ? w1 : w0) * inv;
        int e = ids[p];
        int pos = atomicAdd(&cnt[e], 1);
        pairlist[e * CAP + pos] = p;
    }
}

// f32 [0..7] and [16..23] -> bf16 B fragment (elementwise casts: backend packs
// cvt_pk pairs itself, no shuffles)
static __device__ __forceinline__ v16bf load_bfrag(const float* p) {
    const float4* bp = (const float4*)p;
    float4 q0 = bp[0], q1 = bp[1], q2 = bp[4], q3 = bp[5];
    v16bf r;
    r[0]  = (__bf16)q0.x; r[1]  = (__bf16)q0.y; r[2]  = (__bf16)q0.z; r[3]  = (__bf16)q0.w;
    r[4]  = (__bf16)q1.x; r[5]  = (__bf16)q1.y; r[6]  = (__bf16)q1.z; r[7]  = (__bf16)q1.w;
    r[8]  = (__bf16)q2.x; r[9]  = (__bf16)q2.y; r[10] = (__bf16)q2.z; r[11] = (__bf16)q2.w;
    r[12] = (__bf16)q3.x; r[13] = (__bf16)q3.y; r[14] = (__bf16)q3.z; r[15] = (__bf16)q3.w;
    return r;
}

// ---------------- kernel 2: GEMM1 (x * W1^T) + SiLU*up -> act (bf16) ----------------
// grid: (E * TILES_PER_E, F/256), block: 256 (8 waves)
// WG tile: 32 tokens x (32 gate + 32 up cols) per wave; each B fragment is
// reused by both M-tiles (8 wmma chains per wave).
__global__ __launch_bounds__(256, 1)
void k_gemm1_silu(const float* __restrict__ x, const float* __restrict__ qw,
                  const float* __restrict__ scale,
                  const int* __restrict__ cnt, const int* __restrict__ pairlist,
                  unsigned short* __restrict__ act) {
    int e    = blockIdx.x / TILES_PER_E;
    int tile = blockIdx.x % TILES_PER_E;
    int n = cnt[e];
    if (tile * MROWS >= n) return;

    __shared__ unsigned lds_a[2 * MROWS * 64];   // double-buffered 32x128 bf16 tiles
    __shared__ int      lds_p[MROWS];

    int tid  = threadIdx.x;
    int wave = tid >> 5;
    int lane = tid & 31;
    int half = lane >> 4;
    int lm   = lane & 15;

    if (tid < MROWS) {
        int idx = tile * MROWS + tid;
        lds_p[tid] = (idx < n) ? pairlist[e * CAP + idx] : -1;
    }
    __syncthreads();

    // staging roles: each thread stages two (row, 8-float segment) tasks
    int row0 = tid >> 4;
    int row1 = 16 + (tid >> 4);
    int sseg = tid & 15;
    int sp0 = lds_p[row0];
    int sp1 = lds_p[row1];
    const float* xrow0 = x + (size_t)((sp0 < 0) ? 0 : (sp0 >> 1)) * H;
    const float* xrow1 = x + (size_t)((sp1 < 0) ? 0 : (sp1 >> 1)) * H;

    // wave's output columns: 32 gate cols + matching 32 up cols
    int cg = blockIdx.y * 256 + wave * 32;
    const float* bg_row[2];
    const float* bu_row[2];
#pragma unroll
    for (int j = 0; j < 2; ++j) {
        bg_row[j] = qw + ((size_t)e * 2 * F + (size_t)(cg + j * 16 + lm)) * H;
        bu_row[j] = qw + ((size_t)e * 2 * F + (size_t)(F + cg + j * 16 + lm)) * H;
    }
    const float* sc_e = scale + (size_t)e * (2 * F / BLK) * (H / BLK);
    int obg = cg >> 7;
    int obu = F / BLK + obg;

    v8f accg[2][2] = {{{}, {}}, {{}, {}}};   // [mt][j]
    v8f accu[2][2] = {{{}, {}}, {{}, {}}};

    auto stageA = [&](int kc, int buf) {
        const float4* s0 = (const float4*)(xrow0 + kc + sseg * 8);
        float4 a0 = s0[0], a1 = s0[1];
        uint4 pk;
        pk.x = cvt2(a0.x, a0.y); pk.y = cvt2(a0.z, a0.w);
        pk.z = cvt2(a1.x, a1.y); pk.w = cvt2(a1.z, a1.w);
        *(uint4*)&lds_a[buf * 2048 + row0 * 64 + sseg * 4] = pk;
        const float4* s1 = (const float4*)(xrow1 + kc + sseg * 8);
        float4 b0 = s1[0], b1 = s1[1];
        uint4 pk2;
        pk2.x = cvt2(b0.x, b0.y); pk2.y = cvt2(b0.z, b0.w);
        pk2.z = cvt2(b1.x, b1.y); pk2.w = cvt2(b1.z, b1.w);
        *(uint4*)&lds_a[buf * 2048 + row1 * 64 + sseg * 4] = pk2;
    };

    stageA(0, 0);
    __syncthreads();

    constexpr int NCHUNK = H / 128;
    for (int ci = 0; ci < NCHUNK; ++ci) {
        int kc = ci * 128;
        int buf = ci & 1;

        if (ci + 1 < NCHUNK) {
            stageA(kc + 128, buf ^ 1);
            __builtin_prefetch(bg_row[0] + kc + 128, 0, 0);
            __builtin_prefetch(bu_row[0] + kc + 128, 0, 0);
        }

        v8f tg[2][2] = {{{}, {}}, {{}, {}}};
        v8f tu[2][2] = {{{}, {}}, {{}, {}}};
#pragma unroll
        for (int s = 0; s < 4; ++s) {          // four K=32 sub-steps
            Frag a0, a1;
            int base = buf * 2048 + s * 16 + half * 4;
            a0.q[0] = *(const uint4*)&lds_a[base + lm * 64];
            a0.q[1] = *(const uint4*)&lds_a[base + lm * 64 + 8];
            a1.q[0] = *(const uint4*)&lds_a[base + (16 + lm) * 64];
            a1.q[1] = *(const uint4*)&lds_a[base + (16 + lm) * 64 + 8];

#pragma unroll
            for (int j = 0; j < 2; ++j) {
                v16bf fg = load_bfrag(bg_row[j] + kc + s * 32 + half * 8);
                tg[0][j] = __builtin_amdgcn_wmma_f32_16x16x32_bf16(
                    false, a0.v, false, fg, (short)0, tg[0][j], false, false);
                tg[1][j] = __builtin_amdgcn_wmma_f32_16x16x32_bf16(
                    false, a1.v, false, fg, (short)0, tg[1][j], false, false);
                v16bf fu = load_bfrag(bu_row[j] + kc + s * 32 + half * 8);
                tu[0][j] = __builtin_amdgcn_wmma_f32_16x16x32_bf16(
                    false, a0.v, false, fu, (short)0, tu[0][j], false, false);
                tu[1][j] = __builtin_amdgcn_wmma_f32_16x16x32_bf16(
                    false, a1.v, false, fu, (short)0, tu[1][j], false, false);
            }
        }

        float sg = sc_e[obg * (H / BLK) + ci];
        float su = sc_e[obu * (H / BLK) + ci];
#pragma unroll
        for (int mt = 0; mt < 2; ++mt)
#pragma unroll
            for (int j = 0; j < 2; ++j)
#pragma unroll
                for (int r = 0; r < 8; ++r) {
                    accg[mt][j][r] = fmaf(sg, tg[mt][j][r], accg[mt][j][r]);
                    accu[mt][j][r] = fmaf(su, tu[mt][j][r], accu[mt][j][r]);
                }
        __syncthreads();
    }

    // SiLU(gate) * up -> bf16 via LDS repack (32 x 256 tile = 16KB) for
    // coalesced b128 stores. C/D layout: VGPR r -> M = r + 8*half, N = lm.
    unsigned short* lds16 = (unsigned short*)lds_a;
#pragma unroll
    for (int mt = 0; mt < 2; ++mt)
#pragma unroll
        for (int j = 0; j < 2; ++j)
#pragma unroll
            for (int r = 0; r < 8; ++r) {
                int m = mt * 16 + r + half * 8;
                float g = accg[mt][j][r], u = accu[mt][j][r];
                float a = (g / (1.0f + __expf(-g))) * u;
                lds16[m * 256 + wave * 32 + j * 16 + lm] = f2bf(a);
            }
    __syncthreads();
#pragma unroll
    for (int rep = 0; rep < 2; ++rep) {
        int row = rep ? row1 : row0;
        int p2  = rep ? sp1 : sp0;
        if (p2 >= 0) {
            size_t dst = (size_t)p2 * F + blockIdx.y * 256 + sseg * 16;
            *(uint4*)&act[dst]     = *(const uint4*)&lds16[row * 256 + sseg * 16];
            *(uint4*)&act[dst + 8] = *(const uint4*)&lds16[row * 256 + sseg * 16 + 8];
        }
    }
}

// ---------------- kernel 3: GEMM2 (act * W2^T) + weighted scatter-add ----------------
// grid: (E * TILES_PER_E, H/256), block: 256 (8 waves); WG tile 32 x 256
__global__ __launch_bounds__(256, 1)
void k_gemm2_scatter(const unsigned short* __restrict__ act,
                     const float* __restrict__ qw2, const float* __restrict__ scale2,
                     const int* __restrict__ cnt, const int* __restrict__ pairlist,
                     const float* __restrict__ twn, float* __restrict__ out) {
    int e    = blockIdx.x / TILES_PER_E;
    int tile = blockIdx.x % TILES_PER_E;
    int n = cnt[e];
    if (tile * MROWS >= n) return;

    __shared__ unsigned lds_a[2 * MROWS * 64];
    __shared__ int      lds_p[MROWS];

    int tid  = threadIdx.x;
    int wave = tid >> 5;
    int lane = tid & 31;
    int half = lane >> 4;
    int lm   = lane & 15;

    if (tid < MROWS) {
        int idx = tile * MROWS + tid;
        lds_p[tid] = (idx < n) ? pairlist[e * CAP + idx] : -1;
    }
    __syncthreads();

    int row0 = tid >> 4;
    int row1 = 16 + (tid >> 4);
    int sseg = tid & 15;
    int sp0 = lds_p[row0];
    int sp1 = lds_p[row1];
    const unsigned short* arow0 = act + (size_t)((sp0 < 0) ? 0 : sp0) * F;
    const unsigned short* arow1 = act + (size_t)((sp1 < 0) ? 0 : sp1) * F;

    int hcol = blockIdx.y * 256 + wave * 32;
    const float* brow[2];
#pragma unroll
    for (int j = 0; j < 2; ++j)
        brow[j] = qw2 + ((size_t)e * H + (size_t)(hcol + j * 16 + lm)) * F;
    const float* sc_e = scale2 + (size_t)e * (H / BLK) * (F / BLK);
    int ob = hcol >> 7;

    v8f acc[2][2] = {{{}, {}}, {{}, {}}};   // [mt][j]

    auto stageA = [&](int kc, int buf) {
        *(uint4*)&lds_a[buf * 2048 + row0 * 64 + sseg * 4] =
            *(const uint4*)(arow0 + kc + sseg * 8);
        *(uint4*)&lds_a[buf * 2048 + row1 * 64 + sseg * 4] =
            *(const uint4*)(arow1 + kc + sseg * 8);
    };

    stageA(0, 0);
    __syncthreads();

    constexpr int NCHUNK = F / 128;
    for (int ci = 0; ci < NCHUNK; ++ci) {
        int kc = ci * 128;
        int buf = ci & 1;

        if (ci + 1 < NCHUNK) {
            stageA(kc + 128, buf ^ 1);
            __builtin_prefetch(brow[0] + kc + 128, 0, 0);
        }

        v8f tacc[2][2] = {{{}, {}}, {{}, {}}};
#pragma unroll
        for (int s = 0; s < 4; ++s) {
            Frag a0, a1;
            int base = buf * 2048 + s * 16 + half * 4;
            a0.q[0] = *(const uint4*)&lds_a[base + lm * 64];
            a0.q[1] = *(const uint4*)&lds_a[base + lm * 64 + 8];
            a1.q[0] = *(const uint4*)&lds_a[base + (16 + lm) * 64];
            a1.q[1] = *(const uint4*)&lds_a[base + (16 + lm) * 64 + 8];

#pragma unroll
            for (int j = 0; j < 2; ++j) {
                v16bf fb = load_bfrag(brow[j] + kc + s * 32 + half * 8);
                tacc[0][j] = __builtin_amdgcn_wmma_f32_16x16x32_bf16(
                    false, a0.v, false, fb, (short)0, tacc[0][j], false, false);
                tacc[1][j] = __builtin_amdgcn_wmma_f32_16x16x32_bf16(
                    false, a1.v, false, fb, (short)0, tacc[1][j], false, false);
            }
        }

        float s2 = sc_e[ob * (F / BLK) + ci];
#pragma unroll
        for (int mt = 0; mt < 2; ++mt)
#pragma unroll
            for (int j = 0; j < 2; ++j)
#pragma unroll
                for (int r = 0; r < 8; ++r)
                    acc[mt][j][r] = fmaf(s2, tacc[mt][j][r], acc[mt][j][r]);
        __syncthreads();
    }

#pragma unroll
    for (int mt = 0; mt < 2; ++mt)
#pragma unroll
        for (int r = 0; r < 8; ++r) {
            int m = mt * 16 + r + half * 8;
            int p = lds_p[m];
            if (p >= 0) {
                int t = p >> 1;
                float w = twn[p];
#pragma unroll
                for (int j = 0; j < 2; ++j)
                    atomicAdd(&out[(size_t)t * H + hcol + j * 16 + lm], w * acc[mt][j][r]);
            }
        }
}

// ---------------- host launcher ----------------
extern "C" void kernel_launch(void* const* d_in, const int* in_sizes, int n_in,
                              void* d_out, int out_size, void* d_ws, size_t ws_size,
                              hipStream_t stream) {
    const float* x    = (const float*)d_in[0];   // (T,H)
    const float* tw   = (const float*)d_in[1];   // (T,K)
    const int*   ids  = (const int*)  d_in[2];   // (T,K)
    const float* w1q  = (const float*)d_in[3];   // (E,2F,H)
    const float* w1s  = (const float*)d_in[4];   // (E,2F/128,H/128)
    const float* w2q  = (const float*)d_in[5];   // (E,H,F)
    const float* w2s  = (const float*)d_in[6];   // (E,H/128,F/128)
    float* out = (float*)d_out;                  // (T,H)

    // workspace layout
    char* ws = (char*)d_ws;
    int*   cnt      = (int*)ws;                         // E ints (pad 256B)
    int*   pairlist = (int*)(ws + 256);                 // E*CAP ints
    float* twn      = (float*)(ws + 256 + sizeof(int) * E * CAP);         // T*K floats
    unsigned short* act =
        (unsigned short*)(ws + 256 + sizeof(int) * E * CAP + sizeof(float) * T * TOPK);
    (void)n_in; (void)in_sizes; (void)out_size; (void)ws_size;

    // 0) zero output + counters
    int nOut = T * H;
    k_zero<<<(nOut + 255) / 256, 256, 0, stream>>>(out, nOut, cnt);

    // 1) routing
    k_route<<<(T + 255) / 256, 256, 0, stream>>>(tw, ids, cnt, pairlist, twn);

    // 2) GEMM1 + SiLU
    dim3 g1(E * TILES_PER_E, F / 256);
    k_gemm1_silu<<<g1, 256, 0, stream>>>(x, w1q, w1s, cnt, pairlist, act);

    // 3) GEMM2 + scatter
    dim3 g2(E * TILES_PER_E, H / 256);
    k_gemm2_scatter<<<g2, 256, 0, stream>>>(act, w2q, w2s, cnt, pairlist, twn, out);
}
